// MultiHeadAttention_3143916060975
// MI455X (gfx1250) — matrix operand order repaired and checked by reading, not compiled, not run on hardware
//
#include <hip/hip_runtime.h>

// ---------------------------------------------------------------------------
// MI455X (gfx1250) multi-head attention, f16 WMMA pipeline with async
// global->LDS staging (CDNA5 GLOBAL_LOAD_ASYNC_TO_LDS_B128 + ASYNCcnt).
// ---------------------------------------------------------------------------

typedef _Float16 half_t;
typedef __attribute__((ext_vector_type(16))) _Float16 v16h;
typedef __attribute__((ext_vector_type(8)))  _Float16 v8h;
typedef __attribute__((ext_vector_type(8)))  float    v8f;

#define D_MODEL 1024
#define BATCH   4
#define SEQ     2048
#define NHEAD   16
#define HDIM    64
#define TOKENS  (BATCH * SEQ)

// wave32 split-counter fences (CDNA5)
#define LDS_WAVE_FENCE() asm volatile("s_wait_dscnt 0" ::: "memory")
#define ASYNC_WAIT()     asm volatile("s_wait_asynccnt 0" ::: "memory")

// Per-lane async DMA: 16 bytes global -> LDS, tracked by ASYNCcnt.
// VDST carries the per-lane LDS byte address (generic LDS addr low 32 bits).
__device__ __forceinline__ void async_copy_b128(const half_t* gsrc,
                                                half_t* ldst) {
  unsigned lds_addr = (unsigned)(size_t)ldst;
  unsigned long long gaddr = (unsigned long long)(size_t)gsrc;
  asm volatile("global_load_async_to_lds_b128 %0, %1, off"
               :
               : "v"(lds_addr), "v"(gaddr)
               : "memory");
}

// ---- fragment loaders (layouts per CDNA5 ISA 7.12.2) ------------
// A 16x32 f16: lane m = lane&15, kbase = (lane>>4)*8, element j -> k = kbase + j + (j&8)
__device__ __forceinline__ v16h load_frag_gap(const half_t* p) {
  v8h lo = *(const v8h*)(p);
  v8h hi = *(const v8h*)(p + 16);
  v16h r;
#pragma unroll
  for (int i = 0; i < 8; ++i) { r[i] = lo[i]; r[i + 8] = hi[i]; }
  return r;
}
// B 32x16 f16: lane n = lane&15, kbase = (lane>>4)*8, element j -> k = kbase + j + (j&8)
__device__ __forceinline__ v16h load_frag_contig(const half_t* p) {
  v8h lo = *(const v8h*)(p);
  v8h hi = *(const v8h*)(p + 16);
  v16h r;
#pragma unroll
  for (int i = 0; i < 8; ++i) { r[i] = lo[i]; r[i + 8] = hi[i]; }
  return r;
}

__device__ __forceinline__ v8f wmma_f16(v16h a, v16h b, v8f c) {
  return __builtin_amdgcn_wmma_f32_16x16x32_f16(false, a, false, b, (short)0, c,
                                                false, false);
}

// ---------------------------------------------------------------------------
// f32 -> f16 conversion
// ---------------------------------------------------------------------------
__global__ __launch_bounds__(256) void cvt_f32_f16_kernel(
    const float* __restrict__ in, half_t* __restrict__ out, int n) {
  int i = blockIdx.x * 256 + threadIdx.x;
  if (i < n) out[i] = (half_t)in[i];
}

// ---------------------------------------------------------------------------
// C[M,N] = A[M,K] @ B[K,N]   (all f16 in; f16 out, or f32 out + bias)
// Macro tile 128x64, BK=32, 256 threads = 8 waves, wave w owns rows w*16..+15.
// A tile staged by async DMA; B tile transposed via VGPRs.
// ---------------------------------------------------------------------------
__global__ __launch_bounds__(256) void gemm_wmma_kernel(
    const half_t* __restrict__ A, const half_t* __restrict__ Bw,
    const float* __restrict__ bias, half_t* __restrict__ Ch,
    float* __restrict__ Cf, int M, int N, int K) {
  __shared__ __attribute__((aligned(16))) half_t lA[128 * 32];   // [m][k]
  __shared__ __attribute__((aligned(16))) half_t lBT[64 * 32];   // [n][k]

  const int tid  = threadIdx.x;
  const int lane = tid & 31;
  const int wave = tid >> 5;
  const int m0   = blockIdx.x * 128;
  const int n0   = blockIdx.y * 64;
  const int nloc = lane & 15;
  const int hi8  = (lane >> 4) * 8;

  v8f acc[4];
#pragma unroll
  for (int i = 0; i < 4; ++i) acc[i] = (v8f){0.f,0.f,0.f,0.f,0.f,0.f,0.f,0.f};

  const int am   = tid >> 1;            // A-stage row (0..127)
  const int apt  = (tid & 1) * 16;      // A-stage 16-half part
  const int bk   = tid >> 3;            // B-stage k (0..31)
  const int bn   = (tid & 7) * 8;       // B-stage n0 (0..56)

  for (int k0 = 0; k0 < K; k0 += 32) {
    __syncthreads();
    // stage A 128x32: two async b128 DMAs per thread (no VGPR round-trip)
    {
      const half_t* src = A + (size_t)(m0 + am) * K + k0 + apt;
      async_copy_b128(src,      lA + am * 32 + apt);
      async_copy_b128(src + 8,  lA + am * 32 + apt + 8);
    }
    // stage B 32x64 transposed into [n][k] (element scatter -> VGPR path)
    {
      const half_t* src = Bw + (size_t)(k0 + bk) * N + n0 + bn;
      v8h v = *(const v8h*)src;
#pragma unroll
      for (int i = 0; i < 8; ++i) lBT[(bn + i) * 32 + bk] = v[i];
    }
    // prefetch next k-step tiles into cache while we compute
    if (k0 + 32 < K) {
      __builtin_prefetch(A + (size_t)(m0 + am) * K + k0 + 32 + apt, 0, 1);
      __builtin_prefetch(Bw + (size_t)(k0 + 32 + bk) * N + n0 + bn, 0, 1);
    }
    ASYNC_WAIT();      // ASYNCcnt==0: async DMA landed in LDS
    __syncthreads();   // dscnt flushed + block barrier

    v16h a = load_frag_gap(lA + (wave * 16 + nloc) * 32 + hi8);
    v16h bf[4];
#pragma unroll
    for (int nt = 0; nt < 4; ++nt)
      bf[nt] = load_frag_contig(lBT + (nt * 16 + nloc) * 32 + hi8);
#pragma unroll
    for (int nt = 0; nt < 4; ++nt)
      acc[nt] = wmma_f16(a, bf[nt], acc[nt]);
  }

  // store: C/D layout m = r + 8*(lane>>4), n = lane&15
  const int mrow = m0 + wave * 16 + hi8;
#pragma unroll
  for (int nt = 0; nt < 4; ++nt) {
    int c = n0 + nt * 16 + nloc;
    float bsum = (Cf && bias) ? bias[c] : 0.f;
#pragma unroll
    for (int r = 0; r < 8; ++r) {
      size_t idx = (size_t)(mrow + r) * N + c;
      if (Cf) Cf[idx] = acc[nt][r] + bsum;
      else    Ch[idx] = (half_t)acc[nt][r];
    }
  }
}

// ---------------------------------------------------------------------------
// Causal flash attention. Q/K/V token-major [TOKENS, D_MODEL] f16.
// Block = (q-tile of 128 rows) x (b,h). 8 waves; wave owns 16 query rows.
// K block async-DMA'd to LDS; V^T transposed via VGPRs.
// ---------------------------------------------------------------------------
__global__ __launch_bounds__(256) void flash_attn_kernel(
    const half_t* __restrict__ Q, const half_t* __restrict__ Km,
    const half_t* __restrict__ Vm, half_t* __restrict__ Ctx) {
  __shared__ __attribute__((aligned(16))) half_t lK[32 * 64];    // [key][d]
  __shared__ __attribute__((aligned(16))) half_t lVT[64 * 32];   // [d][key]
  __shared__ __attribute__((aligned(16))) half_t lP[8 * 16 * 32];// per-wave P

  const int tid  = threadIdx.x;
  const int lane = tid & 31;
  const int wave = tid >> 5;
  const int qt   = blockIdx.x;       // 0..SEQ/128-1
  const int bh   = blockIdx.y;       // 0..BATCH*NHEAD-1
  const int b    = bh >> 4;
  const int h    = bh & 15;
  const int nloc = lane & 15;
  const int hi8  = (lane >> 4) * 8;

  const int    qr0     = qt * 128 + wave * 16;  // first query row of wave
  const size_t tokBase = (size_t)b * SEQ;
  const int    dcol    = h * HDIM;

  // Preload Q A-fragments (dh 0..31 and 32..63)
  const half_t* qp = Q + (tokBase + qr0 + nloc) * D_MODEL + dcol + hi8;
  v16h aQ0 = load_frag_gap(qp);
  v16h aQ1 = load_frag_gap(qp + 32);

  v8f acc[4];
#pragma unroll
  for (int i = 0; i < 4; ++i) acc[i] = (v8f){0.f,0.f,0.f,0.f,0.f,0.f,0.f,0.f};
  float m_i[8], l_i[8];
#pragma unroll
  for (int r = 0; r < 8; ++r) { m_i[r] = -3.0e38f; l_i[r] = 0.f; }

  const int nChunksBlk = qt * 4 + 4;            // block-uniform (barrier-safe)
  const int nChunksW   = ((qr0 + 15) >> 5) + 1; // wave-uniform predicate

  const int skey = tid >> 3;          // staging key row (0..31)
  const int sd0  = (tid & 7) * 8;     // staging dh chunk

  for (int c = 0; c < nChunksBlk; ++c) {
    const int kb = c * 32;
    __syncthreads();
    {  // K block: straight copy -> async DMA; V block: transpose via VGPRs
      const half_t* ksrc = Km + (tokBase + kb + skey) * D_MODEL + dcol + sd0;
      async_copy_b128(ksrc, lK + skey * 64 + sd0);
      v8h vv = *(const v8h*)(Vm + (tokBase + kb + skey) * D_MODEL + dcol + sd0);
#pragma unroll
      for (int i = 0; i < 8; ++i) lVT[(sd0 + i) * 32 + skey] = vv[i];
    }
    ASYNC_WAIT();
    __syncthreads();
    if (c >= nChunksW) continue;  // wave-uniform: EXEC stays all-1s inside

    // scores: two 16x16 tiles (keys kb..kb+15, kb+16..kb+31)
    v8f sc0 = (v8f){0.f,0.f,0.f,0.f,0.f,0.f,0.f,0.f};
    v8f sc1 = sc0;
    {
      v16h b00 = load_frag_contig(lK + (nloc) * 64 + 0  + hi8);
      v16h b01 = load_frag_contig(lK + (nloc) * 64 + 32 + hi8);
      v16h b10 = load_frag_contig(lK + (16 + nloc) * 64 + 0  + hi8);
      v16h b11 = load_frag_contig(lK + (16 + nloc) * 64 + 32 + hi8);
      sc0 = wmma_f16(aQ0, b00, sc0);
      sc0 = wmma_f16(aQ1, b01, sc0);
      sc1 = wmma_f16(aQ0, b10, sc1);
      sc1 = wmma_f16(aQ1, b11, sc1);
    }

    // online softmax (rows live across 16 lanes of each half-wave)
    float p0[8], p1[8];
#pragma unroll
    for (int r = 0; r < 8; ++r) {
      const int qrow = qr0 + hi8 + r;
      const int kg0  = kb + nloc;
      const int kg1  = kb + 16 + nloc;
      float s0 = (kg0 <= qrow) ? sc0[r] * 0.125f : -3.0e38f;
      float s1 = (kg1 <= qrow) ? sc1[r] * 0.125f : -3.0e38f;
      float mx = fmaxf(s0, s1);
      mx = fmaxf(mx, __shfl_xor(mx, 1, 32));
      mx = fmaxf(mx, __shfl_xor(mx, 2, 32));
      mx = fmaxf(mx, __shfl_xor(mx, 4, 32));
      mx = fmaxf(mx, __shfl_xor(mx, 8, 32));
      const float mnew = fmaxf(m_i[r], mx);
      const float e0 = (s0 > -1.0e37f) ? __expf(s0 - mnew) : 0.f;
      const float e1 = (s1 > -1.0e37f) ? __expf(s1 - mnew) : 0.f;
      p0[r] = e0; p1[r] = e1;
      float sum = e0 + e1;
      sum += __shfl_xor(sum, 1, 32);
      sum += __shfl_xor(sum, 2, 32);
      sum += __shfl_xor(sum, 4, 32);
      sum += __shfl_xor(sum, 8, 32);
      const float scale = __expf(m_i[r] - mnew);
      l_i[r] = l_i[r] * scale + sum;
      m_i[r] = mnew;
      acc[0][r] *= scale; acc[1][r] *= scale;
      acc[2][r] *= scale; acc[3][r] *= scale;
    }

    // C-layout probs -> per-wave LDS (row-major 16x32) -> A-layout fragment
    half_t* pw = lP + wave * 512;
#pragma unroll
    for (int r = 0; r < 8; ++r) {
      const int mrow = hi8 + r;
      pw[mrow * 32 + nloc]      = (half_t)p0[r];
      pw[mrow * 32 + 16 + nloc] = (half_t)p1[r];
    }
    LDS_WAVE_FENCE();  // same-wave LDS RAW: stores complete before frag reads
    v16h aP = load_frag_gap(pw + nloc * 32 + hi8);

    v16h bV[4];
#pragma unroll
    for (int nt = 0; nt < 4; ++nt)
      bV[nt] = load_frag_contig(lVT + (nt * 16 + nloc) * 32 + hi8);
#pragma unroll
    for (int nt = 0; nt < 4; ++nt)
      acc[nt] = wmma_f16(aP, bV[nt], acc[nt]);
  }

  // normalize and write ctx back token-major at this head's columns
#pragma unroll
  for (int r = 0; r < 8; ++r) {
    const float inv = 1.0f / l_i[r];
    const size_t tok = tokBase + qr0 + hi8 + r;
#pragma unroll
    for (int nt = 0; nt < 4; ++nt)
      Ctx[tok * D_MODEL + dcol + nt * 16 + nloc] = (half_t)(acc[nt][r] * inv);
  }
}

// ---------------------------------------------------------------------------
// Launch
// ---------------------------------------------------------------------------
extern "C" void kernel_launch(void* const* d_in, const int* in_sizes, int n_in,
                              void* d_out, int out_size, void* d_ws,
                              size_t ws_size, hipStream_t stream) {
  const float* x  = (const float*)d_in[0];
  const float* Wq = (const float*)d_in[1];
  const float* Wk = (const float*)d_in[2];
  const float* Wv = (const float*)d_in[3];
  const float* Wo = (const float*)d_in[4];
  const float* bo = (const float*)d_in[5];
  float* out = (float*)d_out;

  const size_t NTOK = (size_t)TOKENS * D_MODEL;  // 8.39M elems
  const size_t NW   = (size_t)D_MODEL * D_MODEL; // 1.05M elems

  half_t* xh  = (half_t*)d_ws;
  half_t* qh  = xh  + NTOK;
  half_t* kh  = qh  + NTOK;
  half_t* vh  = kh  + NTOK;
  half_t* ch  = vh  + NTOK;
  half_t* wqh = ch  + NTOK;
  half_t* wkh = wqh + NW;
  half_t* wvh = wkh + NW;
  half_t* woh = wvh + NW;

  // 1) convert inputs to f16
  cvt_f32_f16_kernel<<<(int)(NTOK / 256), 256, 0, stream>>>(x,  xh,  (int)NTOK);
  cvt_f32_f16_kernel<<<(int)(NW / 256),   256, 0, stream>>>(Wq, wqh, (int)NW);
  cvt_f32_f16_kernel<<<(int)(NW / 256),   256, 0, stream>>>(Wk, wkh, (int)NW);
  cvt_f32_f16_kernel<<<(int)(NW / 256),   256, 0, stream>>>(Wv, wvh, (int)NW);
  cvt_f32_f16_kernel<<<(int)(NW / 256),   256, 0, stream>>>(Wo, woh, (int)NW);

  // 2) QKV projections
  dim3 gg(TOKENS / 128, D_MODEL / 64);
  gemm_wmma_kernel<<<gg, 256, 0, stream>>>(xh, wqh, nullptr, qh, nullptr,
                                           TOKENS, D_MODEL, D_MODEL);
  gemm_wmma_kernel<<<gg, 256, 0, stream>>>(xh, wkh, nullptr, kh, nullptr,
                                           TOKENS, D_MODEL, D_MODEL);
  gemm_wmma_kernel<<<gg, 256, 0, stream>>>(xh, wvh, nullptr, vh, nullptr,
                                           TOKENS, D_MODEL, D_MODEL);

  // 3) causal flash attention
  dim3 fg(SEQ / 128, BATCH * NHEAD);
  flash_attn_kernel<<<fg, 256, 0, stream>>>(qh, kh, vh, ch);

  // 4) output projection with bias (f32 out)
  gemm_wmma_kernel<<<gg, 256, 0, stream>>>(ch, woh, bo, nullptr, out,
                                           TOKENS, D_MODEL, D_MODEL);
}
